// CausalSelfAttention_45449343926632
// MI455X (gfx1250) — matrix operand order repaired and checked by reading, not compiled, not run on hardware
//
#include <hip/hip_runtime.h>

#define NB_   4
#define SEQ_  2048
#define CDIM_ 1024
#define NH_   16
#define DH_   64
#define BN_   128      // GEMM block N tile
#define BK_   32       // GEMM k step

typedef __attribute__((ext_vector_type(16))) __bf16 v16bf;
typedef __attribute__((ext_vector_type(8)))  float  v8f;

__device__ __forceinline__ v8f vzero8() {
  v8f z = {0.f,0.f,0.f,0.f,0.f,0.f,0.f,0.f};
  return z;
}

// A-fragment (16x32 bf16, row-major source, stride lda elements).
// Lane L: rows M=L&15; holds K = koff..koff+7 and koff+16..koff+23, koff=(L>>4)*8.
__device__ __forceinline__ v16bf load_frag_a(const __bf16* p, int lda) {
  const int lane = threadIdx.x & 31;
  const __bf16* base = p + (lane & 15) * lda + ((lane >> 4) << 3);
  union { uint4 u[2]; v16bf v; } r;
  r.u[0] = *(const uint4*)(base);
  r.u[1] = *(const uint4*)(base + 16);
  return r.v;
}

// B-fragment (32x16 bf16) loaded from BT stored row-major [N=16 rows, K=32 cols].
// Lane L: column N=L&15; holds K = kb..kb+7 and kb+16..kb+23, kb=(L>>4)*8.
__device__ __forceinline__ v16bf load_frag_bt(const __bf16* p, int ldb) {
  const int lane = threadIdx.x & 31;
  const __bf16* base = p + (lane & 15) * ldb + ((lane >> 4) << 3);
  union { uint4 u[2]; v16bf v; } r;
  r.u[0] = *(const uint4*)(base);
  r.u[1] = *(const uint4*)(base + 16);
  return r.v;
}

__device__ __forceinline__ v8f wmma_bf16(v16bf a, v16bf b, v8f c) {
  return __builtin_amdgcn_wmma_f32_16x16x32_bf16(false, a, false, b,
                                                 (short)0, c, false, false);
}

// CDNA5 async global->LDS copy (16B per active lane), tracked by ASYNCcnt.
__device__ __forceinline__ void async_g2l_b128(unsigned int lds_off,
                                               const void* gptr) {
  asm volatile("global_load_async_to_lds_b128 %0, %1, off"
               :: "v"(lds_off), "v"((unsigned long long)(uintptr_t)gptr)
               : "memory");
}
__device__ __forceinline__ void wait_async0() {
  asm volatile("s_wait_asynccnt 0x0" ::: "memory");
}

// ---------------- stage 1: fp32 -> bf16 ----------------
__global__ void cvt_f32_bf16_kernel(const float* __restrict__ in,
                                    __bf16* __restrict__ out, int n) {
  int i = blockIdx.x * blockDim.x + threadIdx.x;
  int stride = gridDim.x * blockDim.x;
  for (; i < n; i += stride) out[i] = (__bf16)in[i];
}

__global__ void zero_bf16_kernel(__bf16* __restrict__ p, int n) {
  int i = blockIdx.x * blockDim.x + threadIdx.x;
  if (i < n) p[i] = (__bf16)0.0f;
}

// ---------------- stage 2: QKV GEMM + head scatter ----------------
// X: [B*T, C] bf16.  Wa: [3C, C] bf16 (rows are output features -> BT layout).
// Q,K: [B,H,T,D]; VT: [B,H,D,T].  Q pre-scaled by 1/sqrt(D).
// B tile (128 x 32) staged in LDS via async copies, double-buffered.
__global__ void __launch_bounds__(256)
qkv_gemm_kernel(const __bf16* __restrict__ X, const __bf16* __restrict__ Wa,
                __bf16* __restrict__ Q, __bf16* __restrict__ K,
                __bf16* __restrict__ VT) {
  __shared__ alignas(16) __bf16 Bs[2][BN_ * BK_];   // 2 x 8KB
  const int tid  = threadIdx.x;
  const int lane = tid & 31;
  const int w    = tid >> 5;
  const int wm = w & 3, wn = w >> 2;               // 4 x 2 waves -> 64 x 128 tile
  const int m0   = blockIdx.x * 64 + wm * 16;
  const int nblk = blockIdx.y * BN_;
  const __bf16* arow = X + (size_t)m0 * CDIM_;

  // this thread's two 16B chunks of the 8KB B tile
  const int c0   = tid * 2;
  const int row0 = c0 >> 2, sub0 = (c0 & 3) * 8;       // elem offsets
  const int row1 = (c0 + 1) >> 2, sub1 = ((c0 + 1) & 3) * 8;

  auto issueB = [&](int kk, int buf) {
    const __bf16* g0 = Wa + (size_t)(nblk + row0) * CDIM_ + kk + sub0;
    const __bf16* g1 = Wa + (size_t)(nblk + row1) * CDIM_ + kk + sub1;
    async_g2l_b128((unsigned int)(uintptr_t)&Bs[buf][row0 * BK_ + sub0], g0);
    async_g2l_b128((unsigned int)(uintptr_t)&Bs[buf][row1 * BK_ + sub1], g1);
  };

  v8f acc[4];
#pragma unroll
  for (int t = 0; t < 4; ++t) acc[t] = vzero8();

  issueB(0, 0);
  wait_async0();
  __syncthreads();

  v16bf a = load_frag_a(arow, CDIM_);
  const int nsteps = CDIM_ / BK_;
  for (int ks = 0; ks < nsteps; ++ks) {
    const int buf = ks & 1;
    const bool more = (ks + 1) < nsteps;
    if (more) issueB((ks + 1) * BK_, buf ^ 1);         // overlap copy w/ compute
    v16bf a_cur = a;
    if (more) a = load_frag_a(arow + (ks + 1) * BK_, CDIM_);  // A prefetch
    const __bf16* bs = &Bs[buf][(wn * 64) * BK_];
#pragma unroll
    for (int t = 0; t < 4; ++t) {
      v16bf b = load_frag_bt(bs + t * 16 * BK_, BK_);
      acc[t] = wmma_bf16(a_cur, b, acc[t]);
    }
    wait_async0();
    __syncthreads();
  }

  const int rbase = (lane >> 4) << 3;
#pragma unroll
  for (int t = 0; t < 4; ++t) {
    const int f     = nblk + wn * 64 + t * 16 + (lane & 15);
    const int which = f >> 10;          // 0=q 1=k 2=v  (C=1024)
    const int fc    = f & (CDIM_ - 1);
    const int h     = fc >> 6;          // D=64
    const int d     = fc & (DH_ - 1);
#pragma unroll
    for (int j = 0; j < 8; ++j) {
      const int m  = m0 + rbase + j;
      const int b_ = m >> 11;           // T=2048
      const int tq = m & (SEQ_ - 1);
      const size_t bh = (size_t)(b_ * NH_ + h);
      const float val = acc[t][j];
      if (which == 0)
        Q[(bh * SEQ_ + tq) * DH_ + d] = (__bf16)(val * 0.125f);
      else if (which == 1)
        K[(bh * SEQ_ + tq) * DH_ + d] = (__bf16)val;
      else
        VT[(bh * DH_ + d) * SEQ_ + tq] = (__bf16)val;
    }
  }
}

// ---------------- stage 3: causal flash attention ----------------
// grid = (B*H, T/128); 8 waves, each owns a 16-row query strip.
__global__ void __launch_bounds__(256)
flash_attn_kernel(const __bf16* __restrict__ Q, const __bf16* __restrict__ K,
                  const __bf16* __restrict__ VT, __bf16* __restrict__ Y) {
  __shared__ alignas(16) __bf16 ldsP[8][16][32];   // per-wave P staging
  const int lane  = threadIdx.x & 31;
  const int w     = threadIdx.x >> 5;
  const int bh    = blockIdx.x;
  const int b_    = bh >> 4;
  const int h     = bh & (NH_ - 1);
  const int qw    = blockIdx.y * 128 + w * 16;
  const int rbase = (lane >> 4) << 3;
  const float NI  = -__builtin_inff();

  const __bf16* qptr = Q + ((size_t)bh * SEQ_ + qw) * DH_;
  const v16bf aq0 = load_frag_a(qptr, DH_);        // d = 0..31
  const v16bf aq1 = load_frag_a(qptr + 32, DH_);   // d = 32..63

  v8f accv[4];
  float mrow[8], lrow[8];
#pragma unroll
  for (int t = 0; t < 4; ++t) accv[t] = vzero8();
#pragma unroll
  for (int j = 0; j < 8; ++j) { mrow[j] = NI; lrow[j] = 0.f; }

  __bf16* myP = &ldsP[w][0][0];
  const int kend = qw + 16;                        // causal upper bound

  for (int k0 = 0; k0 < kend; k0 += 32) {
    const __bf16* kptr = K + ((size_t)bh * SEQ_ + k0) * DH_;
    // S tile 0: keys k0..k0+15
    v8f s0 = vzero8();
    s0 = wmma_bf16(aq0, load_frag_bt(kptr, DH_), s0);
    s0 = wmma_bf16(aq1, load_frag_bt(kptr + 32, DH_), s0);
    // S tile 1: keys k0+16..k0+31 (uniform branch: EXEC stays all-ones)
    const bool have2 = (k0 + 16) < kend;
    v8f s1;
    if (have2) {
      const __bf16* kptr2 = kptr + 16 * DH_;
      s1 = vzero8();
      s1 = wmma_bf16(aq0, load_frag_bt(kptr2, DH_), s1);
      s1 = wmma_bf16(aq1, load_frag_bt(kptr2 + 32, DH_), s1);
    } else {
#pragma unroll
      for (int j = 0; j < 8; ++j) s1[j] = NI;
    }
    // causal mask on diagonal-overlapping blocks
    if (k0 + 31 > qw) {
      const int key0 = k0 + (lane & 15);
#pragma unroll
      for (int j = 0; j < 8; ++j) {
        const int qrow = qw + rbase + j;
        if (key0 > qrow)      s0[j] = NI;
        if (key0 + 16 > qrow) s1[j] = NI;
      }
    }
    // online softmax: row stats live in 16-lane halves; butterflies stay in-half
#pragma unroll
    for (int j = 0; j < 8; ++j) {
      float vmax = fmaxf(s0[j], s1[j]);
      vmax = fmaxf(vmax, __shfl_xor(vmax, 1, 32));
      vmax = fmaxf(vmax, __shfl_xor(vmax, 2, 32));
      vmax = fmaxf(vmax, __shfl_xor(vmax, 4, 32));
      vmax = fmaxf(vmax, __shfl_xor(vmax, 8, 32));
      const float nm = fmaxf(mrow[j], vmax);
      const float p0 = __expf(s0[j] - nm);
      const float p1 = __expf(s1[j] - nm);
      float rs = p0 + p1;
      rs += __shfl_xor(rs, 1, 32);
      rs += __shfl_xor(rs, 2, 32);
      rs += __shfl_xor(rs, 4, 32);
      rs += __shfl_xor(rs, 8, 32);
      const float corr = __expf(mrow[j] - nm);
      lrow[j] = lrow[j] * corr + rs;
      mrow[j] = nm;
#pragma unroll
      for (int t = 0; t < 4; ++t) accv[t][j] *= corr;
      const int r = rbase + j;
      myP[r * 32 + (lane & 15)]      = (__bf16)p0;   // keys k0..k0+15
      myP[r * 32 + 16 + (lane & 15)] = (__bf16)p1;   // keys k0+16..k0+31
    }
    // P @ V : A = P (16x32) from LDS; B columns = contiguous VT rows
    const v16bf ap = load_frag_a(myP, 32);
#pragma unroll
    for (int t = 0; t < 4; ++t) {
      const __bf16* vptr = VT + ((size_t)bh * DH_ + t * 16) * SEQ_ + k0;
      accv[t] = wmma_bf16(ap, load_frag_bt(vptr, SEQ_), accv[t]);
    }
  }

  // normalize and store y back in [B,T,C] layout (bf16)
#pragma unroll
  for (int j = 0; j < 8; ++j) {
    const float inv = 1.0f / lrow[j];
    const int qrow  = qw + rbase + j;
    const size_t rowOff = ((size_t)b_ * SEQ_ + qrow) * CDIM_ + h * DH_;
#pragma unroll
    for (int t = 0; t < 4; ++t)
      Y[rowOff + t * 16 + (lane & 15)] = (__bf16)(accv[t][j] * inv);
  }
}

// ---------------- stage 4: output projection ----------------
__global__ void __launch_bounds__(256)
proj_gemm_kernel(const __bf16* __restrict__ Ybf, const __bf16* __restrict__ Wp,
                 float* __restrict__ Out) {
  __shared__ alignas(16) __bf16 Bs[2][BN_ * BK_];
  const int tid  = threadIdx.x;
  const int lane = tid & 31;
  const int w    = tid >> 5;
  const int wm = w & 3, wn = w >> 2;
  const int m0   = blockIdx.x * 64 + wm * 16;
  const int nblk = blockIdx.y * BN_;
  const __bf16* arow = Ybf + (size_t)m0 * CDIM_;

  const int c0   = tid * 2;
  const int row0 = c0 >> 2, sub0 = (c0 & 3) * 8;
  const int row1 = (c0 + 1) >> 2, sub1 = ((c0 + 1) & 3) * 8;

  auto issueB = [&](int kk, int buf) {
    const __bf16* g0 = Wp + (size_t)(nblk + row0) * CDIM_ + kk + sub0;
    const __bf16* g1 = Wp + (size_t)(nblk + row1) * CDIM_ + kk + sub1;
    async_g2l_b128((unsigned int)(uintptr_t)&Bs[buf][row0 * BK_ + sub0], g0);
    async_g2l_b128((unsigned int)(uintptr_t)&Bs[buf][row1 * BK_ + sub1], g1);
  };

  v8f acc[4];
#pragma unroll
  for (int t = 0; t < 4; ++t) acc[t] = vzero8();

  issueB(0, 0);
  wait_async0();
  __syncthreads();

  v16bf a = load_frag_a(arow, CDIM_);
  const int nsteps = CDIM_ / BK_;
  for (int ks = 0; ks < nsteps; ++ks) {
    const int buf = ks & 1;
    const bool more = (ks + 1) < nsteps;
    if (more) issueB((ks + 1) * BK_, buf ^ 1);
    v16bf a_cur = a;
    if (more) a = load_frag_a(arow + (ks + 1) * BK_, CDIM_);
    const __bf16* bs = &Bs[buf][(wn * 64) * BK_];
#pragma unroll
    for (int t = 0; t < 4; ++t) {
      v16bf b = load_frag_bt(bs + t * 16 * BK_, BK_);
      acc[t] = wmma_bf16(a_cur, b, acc[t]);
    }
    wait_async0();
    __syncthreads();
  }

  const int rbase = (lane >> 4) << 3;
#pragma unroll
  for (int t = 0; t < 4; ++t) {
    const int n = nblk + wn * 64 + t * 16 + (lane & 15);
#pragma unroll
    for (int j = 0; j < 8; ++j) {
      const int m = m0 + rbase + j;
      Out[(size_t)m * CDIM_ + n] = acc[t][j];
    }
  }
}

extern "C" void kernel_launch(void* const* d_in, const int* in_sizes, int n_in,
                              void* d_out, int out_size, void* d_ws, size_t ws_size,
                              hipStream_t stream) {
  (void)in_sizes; (void)n_in; (void)out_size; (void)ws_size;
  const float* x      = (const float*)d_in[0];
  const float* w_attn = (const float*)d_in[1];
  const float* w_proj = (const float*)d_in[2];
  float* out = (float*)d_out;

  const size_t XE = (size_t)NB_ * SEQ_ * CDIM_;   // 8,388,608 elements
  size_t off = 0;
  auto take = [&](size_t elems) {
    __bf16* p = (__bf16*)((char*)d_ws + off);
    off += ((elems * 2 + 255) & ~(size_t)255);
    return p;
  };
  __bf16* xy_bf = take(XE);                 // x (stage 2 input), reused as y (stage 3 out)
  __bf16* wa_bf = take((size_t)3 * CDIM_ * CDIM_);
  __bf16* wp_bf = take((size_t)CDIM_ * CDIM_);
  __bf16* q_bf  = take(XE);
  __bf16* k_bf  = take(XE);
  __bf16* vT_bf = take(XE);
  __bf16* pad   = take(2048);               // zeroed slack for flash B-frag overrun

  // stage 1: conversions
  cvt_f32_bf16_kernel<<<2048, 256, 0, stream>>>(x, xy_bf, (int)XE);
  cvt_f32_bf16_kernel<<<2048, 256, 0, stream>>>(w_attn, wa_bf, 3 * CDIM_ * CDIM_);
  cvt_f32_bf16_kernel<<<1024, 256, 0, stream>>>(w_proj, wp_bf, CDIM_ * CDIM_);
  zero_bf16_kernel<<<8, 256, 0, stream>>>(pad, 2048);

  // stage 2: QKV projection (M=8192, N=3072, K=1024)
  {
    dim3 grid((NB_ * SEQ_) / 64, (3 * CDIM_) / BN_);
    qkv_gemm_kernel<<<grid, 256, 0, stream>>>(xy_bf, wa_bf, q_bf, k_bf, vT_bf);
  }
  // stage 3: causal flash attention (y overwrites x buffer)
  {
    dim3 grid(NB_ * NH_, SEQ_ / 128);
    flash_attn_kernel<<<grid, 256, 0, stream>>>(q_bf, k_bf, vT_bf, xy_bf);
  }
  // stage 4: output projection (M=8192, N=1024, K=1024), fp32 out
  {
    dim3 grid((NB_ * SEQ_) / 64, CDIM_ / BN_);
    proj_gemm_kernel<<<grid, 256, 0, stream>>>(xy_bf, wp_bf, out);
  }
}
